// MultiHeadAttention_56504589746463
// MI455X (gfx1250) — matrix operand order repaired and checked by reading, not compiled, not run on hardware
//
#include <hip/hip_runtime.h>
#include <hip/hip_bf16.h>

#define B_  8
#define S_  2048
#define D_  512
#define H_  8
#define DH_ 64

typedef __bf16 bf16;
typedef bf16  v16bf __attribute__((ext_vector_type(16)));
typedef float v8f   __attribute__((ext_vector_type(8)));
typedef unsigned int u32x4 __attribute__((ext_vector_type(4)));
typedef int si4 __attribute__((ext_vector_type(4)));
typedef int si8 __attribute__((ext_vector_type(8)));

union FragAB { v16bf v; u32x4 q[2]; };

__device__ __forceinline__ unsigned short f2bf_rne(float f) {
    unsigned int u = __float_as_uint(f);
    unsigned int r = u + 0x7FFFu + ((u >> 16) & 1u);
    return (unsigned short)(r >> 16);
}

__device__ __forceinline__ v8f wmma_bf16(const FragAB& a, const FragAB& b, v8f c) {
    return __builtin_amdgcn_wmma_f32_16x16x32_bf16(
        false, a.v, false, b.v, (short)0, c, false, false);
}

// --------------------------------------------------------------- TDM (D#)
// 2D tensor_load_to_lds: tile (tile_d0 x tile_d1) of a (tensor_d0 x tensor_d1)
// bf16 tensor with row stride stride0 (all in elements). Inline asm form is
// portable across ROCm7.2 / amdgpu-toolchain (bypasses the clang builtin).
__device__ __forceinline__ void tdm_load_2d_bf16(unsigned lds_addr,
                                                 const void* gaddr,
                                                 unsigned tensor_d0, unsigned tensor_d1,
                                                 unsigned tile_d0,   unsigned tile_d1,
                                                 unsigned stride0) {
    unsigned long long ga = (unsigned long long)gaddr;
    si4 g0;
    g0[0] = 1;                                        // count=1, user descriptor
    g0[1] = (int)lds_addr;                            // LDS byte address
    g0[2] = (int)(unsigned)(ga & 0xFFFFFFFFu);        // global addr [95:64]
    g0[3] = (int)(((unsigned)(ga >> 32) & 0x01FFFFFFu) | (2u << 30)); // type=2
    si8 g1;
    g1[0] = (int)(1u << 16);                          // data_size=1 -> 2 bytes
    g1[1] = (int)(tensor_d0 << 16);                   // tensor_dim0[15:0] @ 63:48
    g1[2] = (int)(((tensor_d0 >> 16) & 0xFFFFu) | (tensor_d1 << 16));
    g1[3] = (int)(((tensor_d1 >> 16) & 0xFFFFu) | (tile_d0 << 16));
    g1[4] = (int)(tile_d1 & 0xFFFFu);                 // tile_dim1; tile_dim2=0
    g1[5] = (int)stride0;                             // dim0 stride [191:160]
    g1[6] = 0;
    g1[7] = 0;
    asm volatile("tensor_load_to_lds %0, %1" :: "s"(g0), "s"(g1) : "memory");
}

// ---------------------------------------------------------------- converts
__global__ void k_f32_to_bf16(const float* __restrict__ in,
                              unsigned short* __restrict__ out, int n) {
    for (int i = blockIdx.x * blockDim.x + threadIdx.x; i < n;
         i += gridDim.x * blockDim.x)
        out[i] = f2bf_rne(in[i]);
}

__global__ void k_zero_f32(float* __restrict__ p, int n) {
    for (int i = blockIdx.x * blockDim.x + threadIdx.x; i < n;
         i += gridDim.x * blockDim.x)
        p[i] = 0.0f;
}

// ---------------------------------------------------------------- QKV GEMM
// C[m,n] = sum_k X[m,k] * W[n,k]; one wave computes a 16x64 strip:
// A-fragment loaded once per k-step and reused across 4 WMMAs.
// mode 0/1: store bf16 as [B,H,S,DH]; mode 2: store transposed [B,H,DH,S]
__global__ void k_qkv_gemm(const unsigned short* __restrict__ Xbf,
                           const unsigned short* __restrict__ Wbf,
                           unsigned short* __restrict__ Out, int mode) {
    const int lane = threadIdx.x & 31;
    const int wid  = blockIdx.x * (blockDim.x >> 5) + (threadIdx.x >> 5);
    const int NT4  = D_ / 64;             // 8 strips of 64 cols
    const int tm   = wid / NT4;           // 0..1023
    const int tn   = wid % NT4;
    const int col  = lane & 15;
    const int half = lane >> 4;

    const unsigned short* arow = Xbf + (size_t)(tm * 16 + col) * D_;
    const unsigned short* brow = Wbf + (size_t)(tn * 64 + col) * D_;

    v8f c0 = {}, c1 = {}, c2 = {}, c3 = {};
    #pragma unroll 2
    for (int k = 0; k < D_; k += 32) {
        FragAB a, b;
        a.q[0] = *(const u32x4*)(arow + k + 16 * half);
        a.q[1] = *(const u32x4*)(arow + k + 16 * half + 8);
        b.q[0] = *(const u32x4*)(brow + k + 16 * half);
        b.q[1] = *(const u32x4*)(brow + k + 16 * half + 8);
        c0 = wmma_bf16(a, b, c0);
        b.q[0] = *(const u32x4*)(brow + 16 * D_ + k + 16 * half);
        b.q[1] = *(const u32x4*)(brow + 16 * D_ + k + 16 * half + 8);
        c1 = wmma_bf16(a, b, c1);
        b.q[0] = *(const u32x4*)(brow + 32 * D_ + k + 16 * half);
        b.q[1] = *(const u32x4*)(brow + 32 * D_ + k + 16 * half + 8);
        c2 = wmma_bf16(a, b, c2);
        b.q[0] = *(const u32x4*)(brow + 48 * D_ + k + 16 * half);
        b.q[1] = *(const u32x4*)(brow + 48 * D_ + k + 16 * half + 8);
        c3 = wmma_bf16(a, b, c3);
    }

    v8f* cs[4] = { &c0, &c1, &c2, &c3 };
    #pragma unroll
    for (int t = 0; t < 4; ++t) {
        const int ng = tn * 64 + t * 16 + col;
        const int h  = ng >> 6, d = ng & (DH_ - 1);
        v8f c = *cs[t];
        #pragma unroll
        for (int j = 0; j < 8; ++j) {
            int m  = tm * 16 + j + 8 * half;
            int bb = m >> 11;
            int s  = m & (S_ - 1);
            size_t idx;
            if (mode == 2) idx = ((size_t)((bb * H_ + h) * DH_ + d)) * S_ + s;
            else           idx = ((size_t)((bb * H_ + h) * S_ + s)) * DH_ + d;
            Out[idx] = f2bf_rne(c[j]);
        }
    }
}

// ---------------------------------------------------------------- attention
// One block = 8 waves on the SAME (b,h); wave wv handles queries qbase+wv*16.
// Per 32-key step, wave 0 TDM-loads K/V tiles into double-buffered LDS
// (TENSORcnt + barrier protocol); all waves consume via ds_load_b128 B-frags.
// Flash online softmax; normalized output column-summed into pooled[b,:].
__global__ void k_attn(const unsigned short* __restrict__ Qh,   // [B,H,S,DH]
                       const unsigned short* __restrict__ Kh,   // [B,H,S,DH]
                       const unsigned short* __restrict__ Vt,   // [B,H,DH,S]
                       const int* __restrict__ valid,
                       float* __restrict__ pooled) {            // [B,D]
    __shared__ __align__(16) unsigned short Kls[2][32 * DH_];   // [s=32][d=64]
    __shared__ __align__(16) unsigned short Vls[2][DH_ * 32];   // [d=64][s=32]
    __shared__ __align__(16) unsigned short Pst[8][16 * 32];

    const int lane = threadIdx.x & 31;
    const int wv   = threadIdx.x >> 5;
    const int qblk = blockIdx.x & 15;            // 16 q-blocks of 128
    const int h    = (blockIdx.x >> 4) & 7;
    const int b    = blockIdx.x >> 7;
    const int col  = lane & 15;
    const int half = lane >> 4;
    const int vlen = valid[b];

    const size_t bh = (size_t)(b * H_ + h);
    const unsigned short* Kbase = Kh + bh * S_ * DH_;
    const unsigned short* Vbase = Vt + bh * DH_ * S_;

    // resident Q A-fragments (d 0..31 and 32..63)
    const int q0 = qblk * 128 + wv * 16;
    const unsigned short* qrow = Qh + (bh * S_ + q0 + col) * DH_;
    FragAB qa0, qa1;
    qa0.q[0] = *(const u32x4*)(qrow + 16 * half);
    qa0.q[1] = *(const u32x4*)(qrow + 16 * half + 8);
    qa1.q[0] = *(const u32x4*)(qrow + 32 + 16 * half);
    qa1.q[1] = *(const u32x4*)(qrow + 32 + 16 * half + 8);

    v8f o0 = {}, o1 = {}, o2 = {}, o3 = {};
    float mj[8], lj[8];
    #pragma unroll
    for (int j = 0; j < 8; ++j) { mj[j] = -3.0e38f; lj[j] = 0.0f; }

    unsigned short* pbuf = &Pst[wv][0];

    // prologue: stage tiles for kt=0 into buffer 0
    if (wv == 0) {
        tdm_load_2d_bf16((unsigned)(unsigned long long)(const void*)&Kls[0][0],
                         Kbase, DH_, S_, DH_, 32, DH_);
        tdm_load_2d_bf16((unsigned)(unsigned long long)(const void*)&Vls[0][0],
                         Vbase, S_, DH_, 32, DH_, S_);
    }

    for (int kt = 0; kt < S_; kt += 32) {
        const int cur = (kt >> 5) & 1;
        if (wv == 0) __builtin_amdgcn_s_wait_tensorcnt(0);
        __syncthreads();                         // buffer `cur` ready for all
        if (wv == 0 && (kt + 32) < S_) {         // prefetch next into cur^1
            tdm_load_2d_bf16((unsigned)(unsigned long long)(const void*)&Kls[cur ^ 1][0],
                             Kbase + (size_t)(kt + 32) * DH_, DH_, S_, DH_, 32, DH_);
            tdm_load_2d_bf16((unsigned)(unsigned long long)(const void*)&Vls[cur ^ 1][0],
                             Vbase + (kt + 32), S_, DH_, 32, DH_, S_);
        }

        // ---- scores from LDS K tile: two 16x16 key tiles
        v8f c0 = {}, c1 = {};
        {
            const unsigned short* kr0 = &Kls[cur][0] + (size_t)col * DH_;
            const unsigned short* kr1 = kr0 + 16 * DH_;
            FragAB kb;
            kb.q[0] = *(const u32x4*)(kr0 + 16 * half);
            kb.q[1] = *(const u32x4*)(kr0 + 16 * half + 8);
            c0 = wmma_bf16(qa0, kb, c0);
            kb.q[0] = *(const u32x4*)(kr0 + 32 + 16 * half);
            kb.q[1] = *(const u32x4*)(kr0 + 32 + 16 * half + 8);
            c0 = wmma_bf16(qa1, kb, c0);
            kb.q[0] = *(const u32x4*)(kr1 + 16 * half);
            kb.q[1] = *(const u32x4*)(kr1 + 16 * half + 8);
            c1 = wmma_bf16(qa0, kb, c1);
            kb.q[0] = *(const u32x4*)(kr1 + 32 + 16 * half);
            kb.q[1] = *(const u32x4*)(kr1 + 32 + 16 * half + 8);
            c1 = wmma_bf16(qa1, kb, c1);
        }

        // ---- scale + key mask (reference: scores/8, where(mask, ., -1e6))
        const bool v0 = (kt + col) < vlen;
        const bool v1 = (kt + 16 + col) < vlen;
        float s0[8], s1[8], nm[8];
        #pragma unroll
        for (int j = 0; j < 8; ++j) {
            float a0 = v0 ? c0[j] * 0.125f : -1.0e6f;
            float a1 = v1 ? c1[j] * 0.125f : -1.0e6f;
            s0[j] = a0; s1[j] = a1;
            float t = fmaxf(a0, a1);
            t = fmaxf(t, __shfl_xor(t, 1, 32));
            t = fmaxf(t, __shfl_xor(t, 2, 32));
            t = fmaxf(t, __shfl_xor(t, 4, 32));
            t = fmaxf(t, __shfl_xor(t, 8, 32));
            nm[j] = fmaxf(mj[j], t);
        }

        // ---- online softmax + stage P tile to this wave's LDS slab
        #pragma unroll
        for (int j = 0; j < 8; ++j) {
            float p0 = __expf(s0[j] - nm[j]);
            float p1 = __expf(s1[j] - nm[j]);
            float rs = p0 + p1;
            rs += __shfl_xor(rs, 1, 32);
            rs += __shfl_xor(rs, 2, 32);
            rs += __shfl_xor(rs, 4, 32);
            rs += __shfl_xor(rs, 8, 32);
            float sc = __expf(mj[j] - nm[j]);
            lj[j] = lj[j] * sc + rs;
            mj[j] = nm[j];
            o0[j] *= sc; o1[j] *= sc; o2[j] *= sc; o3[j] *= sc;
            int row = j + 8 * half;
            pbuf[row * 32 + col]      = f2bf_rne(p0);
            pbuf[row * 32 + 16 + col] = f2bf_rne(p1);
        }
        asm volatile("s_wait_dscnt 0" ::: "memory");  // same-wave LDS ordering

        FragAB pa;                                    // P as 16x32 A-fragment
        pa.q[0] = *(const u32x4*)(pbuf + col * 32 + 16 * half);
        pa.q[1] = *(const u32x4*)(pbuf + col * 32 + 16 * half + 8);

        // ---- O += P x V from LDS V tile ([d][s] rows of 32)
        const unsigned short* vb0 = &Vls[cur][0] + (size_t)col * 32 + 16 * half;
        FragAB vb;
        vb.q[0] = *(const u32x4*)(vb0);
        vb.q[1] = *(const u32x4*)(vb0 + 8);
        o0 = wmma_bf16(pa, vb, o0);
        vb.q[0] = *(const u32x4*)(vb0 + 16 * 32);
        vb.q[1] = *(const u32x4*)(vb0 + 16 * 32 + 8);
        o1 = wmma_bf16(pa, vb, o1);
        vb.q[0] = *(const u32x4*)(vb0 + 32 * 32);
        vb.q[1] = *(const u32x4*)(vb0 + 32 * 32 + 8);
        o2 = wmma_bf16(pa, vb, o2);
        vb.q[0] = *(const u32x4*)(vb0 + 48 * 32);
        vb.q[1] = *(const u32x4*)(vb0 + 48 * 32 + 8);
        o3 = wmma_bf16(pa, vb, o3);
    }

    // ---- normalize by l, column-sum over 16 query rows, accumulate
    float cs0 = 0.f, cs1 = 0.f, cs2 = 0.f, cs3 = 0.f;
    #pragma unroll
    for (int j = 0; j < 8; ++j) {
        float inv = 1.0f / lj[j];
        cs0 += o0[j] * inv;
        cs1 += o1[j] * inv;
        cs2 += o2[j] * inv;
        cs3 += o3[j] * inv;
    }
    cs0 += __shfl_xor(cs0, 16, 32);
    cs1 += __shfl_xor(cs1, 16, 32);
    cs2 += __shfl_xor(cs2, 16, 32);
    cs3 += __shfl_xor(cs3, 16, 32);
    if (lane < 16) {
        float* pb = pooled + (size_t)b * D_ + h * DH_;
        atomicAdd(pb + 0 * 16 + col, cs0);
        atomicAdd(pb + 1 * 16 + col, cs1);
        atomicAdd(pb + 2 * 16 + col, cs2);
        atomicAdd(pb + 3 * 16 + col, cs3);
    }
}

// ---------------------------------------------------------------- epilogue
__global__ void k_final(const float* __restrict__ pooled,
                        const float* __restrict__ Wo,
                        const float* __restrict__ Wc,
                        const float* __restrict__ bc,
                        float* __restrict__ out) {
    __shared__ float po[D_];
    __shared__ float r0[256], r1[256];
    const int b = blockIdx.x, t = threadIdx.x;
    const float* pa = pooled + (size_t)b * D_;
    for (int d = t; d < D_; d += 256) {
        float acc = 0.0f;
        for (int k = 0; k < D_; ++k) acc += pa[k] * Wo[(size_t)d * D_ + k];
        po[d] = acc * (1.0f / (float)S_);
    }
    __syncthreads();
    float p0 = 0.f, p1 = 0.f;
    for (int d = t; d < D_; d += 256) {
        p0 += po[d] * Wc[d];
        p1 += po[d] * Wc[D_ + d];
    }
    r0[t] = p0; r1[t] = p1;
    __syncthreads();
    for (int s = 128; s > 0; s >>= 1) {
        if (t < s) { r0[t] += r0[t + s]; r1[t] += r1[t + s]; }
        __syncthreads();
    }
    if (t == 0) {
        float l0 = r0[0] + bc[0], l1 = r1[0] + bc[1];
        float m  = fmaxf(l0, l1);
        float ls = logf(__expf(l0 - m) + __expf(l1 - m)) + m;
        out[b * 2 + 0] = l0 - ls;
        out[b * 2 + 1] = l1 - ls;
    }
}

// ---------------------------------------------------------------- launcher
extern "C" void kernel_launch(void* const* d_in, const int* in_sizes, int n_in,
                              void* d_out, int out_size, void* d_ws, size_t ws_size,
                              hipStream_t stream) {
    const float* x  = (const float*)d_in[0];
    const int*   vl = (const int*)d_in[3];
    const float* Wq = (const float*)d_in[4];
    const float* Wk = (const float*)d_in[5];
    const float* Wv = (const float*)d_in[6];
    const float* Wo = (const float*)d_in[7];
    const float* Wc = (const float*)d_in[8];
    const float* bc = (const float*)d_in[9];

    const size_t NX = (size_t)B_ * S_ * D_;
    const size_t NW = (size_t)D_ * D_;

    char* p = (char*)d_ws;
    unsigned short* Xbf = (unsigned short*)p; p += NX * 2;
    unsigned short* Qh  = (unsigned short*)p; p += NX * 2;
    unsigned short* Kh  = (unsigned short*)p; p += NX * 2;
    unsigned short* Vt  = (unsigned short*)p; p += NX * 2;
    unsigned short* Wqb = (unsigned short*)p; p += NW * 2;
    unsigned short* Wkb = (unsigned short*)p; p += NW * 2;
    unsigned short* Wvb = (unsigned short*)p; p += NW * 2;
    float* pooled = (float*)p;

    k_f32_to_bf16<<<2048, 256, 0, stream>>>(x,  Xbf, (int)NX);
    k_f32_to_bf16<<<256,  256, 0, stream>>>(Wq, Wqb, (int)NW);
    k_f32_to_bf16<<<256,  256, 0, stream>>>(Wk, Wkb, (int)NW);
    k_f32_to_bf16<<<256,  256, 0, stream>>>(Wv, Wvb, (int)NW);
    k_zero_f32<<<16, 256, 0, stream>>>(pooled, B_ * D_);

    // (M/16)*(N/64) = 1024*8 = 8192 waves -> 1024 blocks of 8 waves
    k_qkv_gemm<<<1024, 256, 0, stream>>>(Xbf, Wqb, Qh, 0);
    k_qkv_gemm<<<1024, 256, 0, stream>>>(Xbf, Wkb, Kh, 1);
    k_qkv_gemm<<<1024, 256, 0, stream>>>(Xbf, Wvb, Vt, 2);

    // B*H*(S/128) = 1024 blocks; 8 waves share LDS-staged K/V tiles
    k_attn<<<1024, 256, 0, stream>>>(Qh, Kh, Vt, vl, pooled);

    k_final<<<B_, 256, 0, stream>>>(pooled, Wo, Wc, bc, (float*)d_out);
}
